// RBMSites_1726576853263
// MI455X (gfx1250) — matrix operand order repaired and checked by reading, not compiled, not run on hardware
//
#include <hip/hip_runtime.h>
#include <math.h>

typedef __attribute__((ext_vector_type(16))) _Float16 v16h;
typedef __attribute__((ext_vector_type(8)))  float    v8f;
typedef __attribute__((ext_vector_type(4)))  float    v4f;
typedef __attribute__((ext_vector_type(4)))  int      v4i;

#define SORB   64
#define WLEN   1056
#define NBATCH 131072
#define NSTEP  32

// d_ws layout:
//   [0, 256KB)      : Bpk  _Float16[32 m][2 chunk][4 nt][32 lane][16 elem]
//   [256KB, 320KB)  : trig float  [32 m][64 h][8] = {cosA,sinA,cosB,sinB,cosAB,sinAB,0,0}
#define BPK_BYTES  (32u * 2u * 4u * 32u * 16u * 2u)   // 262144

// ---------------------------------------------------------------------------
// Prep 1: repack triangular f32 weights into the f16 WMMA B layout.
// ---------------------------------------------------------------------------
__global__ void repack_B(const float* __restrict__ W, _Float16* __restrict__ Bpk) {
    int tid  = blockIdx.x * blockDim.x + threadIdx.x;   // 131072 elems
    int m    = tid >> 12;
    int rem  = tid & 4095;
    int ch   = rem >> 11;
    int nt   = (rem >> 9) & 3;
    int lane = (rem >> 4) & 31;
    int e    = rem & 15;
    int h    = nt * 16 + (lane & 15);
    int Kg   = ch * 32 + ((e >> 3) << 4) + (lane >> 4) * 8 + (e & 7);
    float v  = 0.0f;
    if (Kg < 2 * m) v = W[h * WLEN + m * (m + 1) + Kg];
    Bpk[tid] = (_Float16)v;
}

// ---------------------------------------------------------------------------
// Prep 2: per-(m,h) trig table for the angle-addition trick.
// ---------------------------------------------------------------------------
__global__ void build_trig(const float* __restrict__ W, float* __restrict__ trig) {
    int tid = blockIdx.x * blockDim.x + threadIdx.x;    // 2048
    int m = tid >> 6, h = tid & 63;
    int start = m * (m + 1);
    float a = W[h * WLEN + start + 2 * m];
    float b = W[h * WLEN + start + 2 * m + 1];
    float sa, ca, sb, cb, sab, cab;
    __sincosf(a, &sa, &ca);
    __sincosf(b, &sb, &cb);
    __sincosf(a + b, &sab, &cab);
    float* o = trig + (size_t)tid * 8;
    o[0] = ca; o[1] = sa; o[2] = cb; o[3] = sb; o[4] = cab; o[5] = sab;
    o[6] = 0.f; o[7] = 0.f;
}

// ---------------------------------------------------------------------------
// Main kernel: one wave32 owns a 16-row batch tile. Phased m-loop:
//   m=0 (no MMA), m=1..16 (1 K-chunk), m=17..31 (2 K-chunks).
// All B vectors for a step are loaded before the first WMMA so the
// 8-load clause pipelines instead of wait-0 before every MMA.
// ---------------------------------------------------------------------------
__global__ void __launch_bounds__(256)
rbm_forward(const int* __restrict__ x, const float* __restrict__ hb,
            const _Float16* __restrict__ Bpk, const float* __restrict__ trig,
            float* __restrict__ out) {
    const int lane = threadIdx.x & 31;
    const int wave = threadIdx.x >> 5;
    const int tile = blockIdx.x * 8 + wave;
    const int base = tile * 16;

    // ---- Build A (x01 as f16, 16x64, split into two 16x32 A-matrices) ----
    const int row = lane & 15;
    const int h0  = (lane >> 4) * 8;
    const int* xr = x + (size_t)(base + row) * SORB;
    v16h a0, a1;
#pragma unroll
    for (int c2 = 0; c2 < 4; ++c2) {
        const v4i* p = (const v4i*)(xr + h0 + c2 * 16);
        v4i q0 = p[0], q1 = p[1];
        int vals[8] = {q0[0], q0[1], q0[2], q0[3], q1[0], q1[1], q1[2], q1[3]};
#pragma unroll
        for (int i = 0; i < 8; ++i) {
            _Float16 v = (vals[i] > 0) ? (_Float16)1.0f : (_Float16)0.0f;
            if (c2 < 2) a0[c2 * 8 + i] = v;
            else        a1[(c2 - 2) * 8 + i] = v;
        }
    }

    // ---- Pack this lane's bookkeeping row into a 64-bit occupation mask ----
    const int bb = (lane >> 4) * 8 + (lane & 7);
    const int* xb = x + (size_t)(base + bb) * SORB;
    unsigned long long bits = 0ull;
#pragma unroll
    for (int c2 = 0; c2 < 16; ++c2) {
        v4i q = ((const v4i*)xb)[c2];
#pragma unroll
        for (int i = 0; i < 4; ++i)
            bits |= ((unsigned long long)(q[i] > 0 ? 1u : 0u)) << (c2 * 4 + i);
    }

    float biasv[4];
#pragma unroll
    for (int nt = 0; nt < 4; ++nt) biasv[nt] = hb[nt * 16 + (lane & 15)];

    float prob = 1.0f;
    int nup = 0, ndn = 0;
    v8f c[4];

    // ---- per-step tail: cos-products, butterfly, mask/normalize/select ----
    auto tail = [&](int m) {
        const int k = 2 * m;
        float tc[4][6];
#pragma unroll
        for (int nt = 0; nt < 4; ++nt) {
            const float* tp = trig + ((size_t)m * 64 + nt * 16 + (lane & 15)) * 8;
            v4f t0 = *(const v4f*)tp;
            tc[nt][0] = t0[0]; tc[nt][1] = t0[1];
            tc[nt][2] = t0[2]; tc[nt][3] = t0[3];
            tc[nt][4] = tp[4]; tc[nt][5] = tp[5];
        }
        float keep0 = 0.f, keep1 = 0.f, keep2 = 0.f, keep3 = 0.f;
#pragma unroll
        for (int r = 0; r < 8; ++r) {
            float p0 = 1.f, p1 = 1.f, p2 = 1.f, p3 = 1.f;
#pragma unroll
            for (int nt = 0; nt < 4; ++nt) {
                float t = c[nt][r] + biasv[nt];
                float st, ct;
                __sincosf(t, &st, &ct);
                ct *= 2.0f; st *= 2.0f;
                p0 *= ct;
                p1 *= ct * tc[nt][0] - st * tc[nt][1];
                p2 *= ct * tc[nt][2] - st * tc[nt][3];
                p3 *= ct * tc[nt][4] - st * tc[nt][5];
            }
#pragma unroll
            for (int s = 1; s < 16; s <<= 1) {
                p0 *= __shfl_xor(p0, s, 32);
                p1 *= __shfl_xor(p1, s, 32);
                p2 *= __shfl_xor(p2, s, 32);
                p3 *= __shfl_xor(p3, s, 32);
            }
            if ((lane & 15) == r) { keep0 = p0; keep1 = p1; keep2 = p2; keep3 = p3; }
        }
        int bit0 = (int)((bits >> k) & 1ull);
        int bit1 = (int)((bits >> (k + 1)) & 1ull);
        float m0 = 1.f, m1 = 1.f, m2 = 1.f, m3 = 1.f;
        if (k >= 16) {
            int lo = m - 16;
            bool occ0 = nup < 16, un0 = nup > lo;
            bool occ1 = ndn < 16, un1 = ndn > lo;
            m0 = (un0  && un1)  ? 1.f : 0.f;
            m1 = (occ0 && un1)  ? 1.f : 0.f;
            m2 = (un0  && occ1) ? 1.f : 0.f;
            m3 = (occ0 && occ1) ? 1.f : 0.f;
        }
        float w0 = keep0 * m0, w1 = keep1 * m1, w2 = keep2 * m2, w3 = keep3 * m3;
        float ss  = w0 * w0 + w1 * w1 + w2 * w2 + w3 * w3;
        float inv = __builtin_amdgcn_rsqf(fmaxf(ss, 1e-24f));   // v_rsq_f32
        int idx = bit0 + 2 * bit1;
        float sel = (idx == 0) ? w0 : (idx == 1) ? w1 : (idx == 2) ? w2 : w3;
        prob *= sel * inv;
        nup += bit0; ndn += bit1;
    };

    const v8f vzero = {0.f, 0.f, 0.f, 0.f, 0.f, 0.f, 0.f, 0.f};

    // ---- m = 0 : theta_c == 0, no MMA ----
    {
        const char* nb = (const char*)Bpk + 8192 + (size_t)lane * 128;
        __builtin_prefetch(nb, 0, 3);
        __builtin_prefetch(nb + 4096, 0, 3);
        c[0] = vzero; c[1] = vzero; c[2] = vzero; c[3] = vzero;
        tail(0);
    }

    // ---- m = 1..16 : one 32-wide K chunk ----
    for (int m = 1; m <= 16; ++m) {
        const _Float16* bbase = Bpk + (size_t)m * 4096 + (size_t)lane * 16;
        v16h b0[4];
#pragma unroll
        for (int nt = 0; nt < 4; ++nt) b0[nt] = *(const v16h*)(bbase + nt * 512);
#pragma unroll
        for (int nt = 0; nt < 4; ++nt)
            c[nt] = __builtin_amdgcn_wmma_f32_16x16x32_f16(
                false, a0, false, b0[nt], (short)0, vzero, false, false);
        // warm next step's B block while the trig section runs
        const char* nb = (const char*)Bpk + (size_t)(m + 1) * 8192 + (size_t)lane * 128;
        __builtin_prefetch(nb, 0, 3);
        __builtin_prefetch(nb + 4096, 0, 3);
        tail(m);
    }

    // ---- m = 17..31 : two 32-wide K chunks ----
    for (int m = 17; m < NSTEP; ++m) {
        const _Float16* bbase = Bpk + (size_t)m * 4096 + (size_t)lane * 16;
        v16h b0[4], b1[4];
#pragma unroll
        for (int nt = 0; nt < 4; ++nt) {
            b0[nt] = *(const v16h*)(bbase + nt * 512);
            b1[nt] = *(const v16h*)(bbase + 2048 + nt * 512);
        }
#pragma unroll
        for (int nt = 0; nt < 4; ++nt) {
            v8f acc = __builtin_amdgcn_wmma_f32_16x16x32_f16(
                false, a0, false, b0[nt], (short)0, vzero, false, false);
            c[nt] = __builtin_amdgcn_wmma_f32_16x16x32_f16(
                false, a1, false, b1[nt], (short)0, acc, false, false);
        }
        if (m + 1 < NSTEP) {
            const char* nb = (const char*)Bpk + (size_t)(m + 1) * 8192 + (size_t)lane * 128;
            __builtin_prefetch(nb, 0, 3);
            __builtin_prefetch(nb + 4096, 0, 3);
        }
        tail(m);
    }

    if ((lane & 15) < 8) out[base + bb] = prob;
}

extern "C" void kernel_launch(void* const* d_in, const int* in_sizes, int n_in,
                              void* d_out, int out_size, void* d_ws, size_t ws_size,
                              hipStream_t stream) {
    const int*   x  = (const int*)d_in[0];
    const float* W  = (const float*)d_in[1];
    const float* hb = (const float*)d_in[2];
    _Float16* Bpk  = (_Float16*)d_ws;
    float*    trig = (float*)((char*)d_ws + BPK_BYTES);

    repack_B  <<<512, 256, 0, stream>>>(W, Bpk);
    build_trig<<<8,   256, 0, stream>>>(W, trig);
    rbm_forward<<<NBATCH / (8 * 16), 256, 0, stream>>>(x, hb, Bpk, trig, (float*)d_out);
}
